// MultilayerLSTM_31129922962130
// MI455X (gfx1250) — matrix operand order repaired and checked by reading, not compiled, not run on hardware
//
#include <hip/hip_runtime.h>
#include <hip/hip_bf16.h>
#include <stdint.h>

// ---------------------------------------------------------------------------
// 2-layer LSTM LM on CDNA5 (gfx1250) using v_wmma_f32_16x16x32_bf16.
// - Weights converted once per launch to bf16, transposed to N-major so the
//   WMMA B-operand (lane = col, K 8*kh..+7 and 16+8*kh..+7 per lane) is two 16-byte loads.
// - All embeddings pre-gathered to bf16 once per launch (off critical path),
//   so the per-timestep kernels have branchless, pure-bf16 K-loops.
// - h-state bf16 (ping-pong buffers), c-state fp32.
// ---------------------------------------------------------------------------

static constexpr int kHID    = 1024;
static constexpr int kEMB    = 512;
static constexpr int kBATCH  = 64;
static constexpr int kSEQ    = 256;
static constexpr int kNCLASS = 32000;

typedef __attribute__((ext_vector_type(16))) __bf16        v16bf;
typedef __attribute__((ext_vector_type(8)))  float         v8f;
typedef __attribute__((ext_vector_type(4)))  unsigned int  u32x4;
typedef __attribute__((ext_vector_type(4)))  float         f32x4;

union Frag {
  v16bf          v;
  u32x4          q[2];
  unsigned short u[16];
};

__device__ __forceinline__ unsigned short f2bf(float f) {
  unsigned int u = __float_as_uint(f);
  u += 0x7FFFu + ((u >> 16) & 1u);   // round-to-nearest-even
  return (unsigned short)(u >> 16);
}

__device__ __forceinline__ float sigmoidf_(float x) {
  return 1.0f / (1.0f + expf(-x));
}

// ---------------------------------------------------------------------------
// Transpose fp32 (K rows x N cols, row-major) -> bf16 (N rows x K cols).
// All dims are multiples of 32 here, so no bounds checks.
// block (32,8), grid (N/32, K/32)
// ---------------------------------------------------------------------------
__global__ __launch_bounds__(256) void transpose_f32_to_bf16(
    const float* __restrict__ src, unsigned short* __restrict__ dst,
    int K, int N) {
  __shared__ float tile[32][33];
  const int tx = threadIdx.x, ty = threadIdx.y;
  const int nb = blockIdx.x * 32, kb = blockIdx.y * 32;
#pragma unroll
  for (int i = 0; i < 4; ++i)
    tile[ty + i * 8][tx] = src[(size_t)(kb + ty + i * 8) * N + (nb + tx)];
  __syncthreads();
#pragma unroll
  for (int i = 0; i < 4; ++i)
    dst[(size_t)(nb + ty + i * 8) * K + (kb + tx)] = f2bf(tile[tx][ty + i * 8]);
}

// ---------------------------------------------------------------------------
// Pre-gather all embeddings to bf16:  xs[t][b][0:512] = bf16(C_emb[X[b,t]]).
// grid = SEQ*BATCH blocks (one row each), block = 128 (4 elems/thread).
// ---------------------------------------------------------------------------
__global__ __launch_bounds__(128) void embed_gather_bf16(
    const float* __restrict__ C_emb, const int* __restrict__ X,
    unsigned short* __restrict__ xs) {
  const int t = blockIdx.x & (kSEQ - 1);
  const int b = blockIdx.x >> 8;   // kSEQ == 256
  const int xid = X[b * kSEQ + t];
  const float* src = C_emb + (size_t)xid * kEMB;
  unsigned short* dst = xs + ((size_t)t * kBATCH + b) * kEMB;
  const int e = threadIdx.x * 4;   // 128 threads x 4 = 512
  f32x4 f = *(const f32x4*)(src + e);
  dst[e + 0] = f2bf(f.x);
  dst[e + 1] = f2bf(f.y);
  dst[e + 2] = f2bf(f.z);
  dst[e + 3] = f2bf(f.w);
}

// ---------------------------------------------------------------------------
// Zero initial states (the parity-1 h buffers are the ones read at t=0).
// ---------------------------------------------------------------------------
__global__ void init_state(unsigned short* h0b, unsigned short* h1b,
                           float* c0, float* c1) {
  int i = blockIdx.x * blockDim.x + threadIdx.x;
  if (i < kBATCH * kHID) {
    h0b[i] = 0; h1b[i] = 0; c0[i] = 0.0f; c1[i] = 0.0f;
  }
}

// ---------------------------------------------------------------------------
// One LSTM layer for one timestep.
//   A = cat(h_prev, x)  (64 x K),  B = Wt[gate]  (N-major bf16, [4][1024][K])
//   K 0..1023   -> recurrent h_prev (bf16, stride 1024)
//   K 1024..K-1 -> x (bf16, stride xdim): xs row (layer 0) or h0_cur (layer 1)
// grid = 64 (16 cols each), block = 256 (8 waves): wave = (gate g, M-tile pair)
// Two branchless K-loops, then fused cell update via LDS.
// ---------------------------------------------------------------------------
__global__ __launch_bounds__(256) void lstm_step_kernel(
    const unsigned short* __restrict__ Wt, int K,
    const unsigned short* __restrict__ h_prev,   // 64x1024 bf16
    const unsigned short* __restrict__ x, int xdim,
    const float* __restrict__ bi, const float* __restrict__ bc,
    const float* __restrict__ bff, const float* __restrict__ bo,
    float* __restrict__ c_state, unsigned short* __restrict__ h_out) {
  __shared__ float gbuf[4][64][16];

  const int tid  = threadIdx.x;
  const int wave = tid >> 5;
  const int lane = tid & 31;
  const int g    = wave & 3;        // gate: 0=i 1=c 2=f 3=o
  const int half = wave >> 2;       // M-tile pair: rows [half*32, half*32+32)
  const int j0   = blockIdx.x * 16; // output column block
  const int lr   = lane & 15;
  const int kh   = lane >> 4;

  v8f acc0 = {};
  v8f acc1 = {};

  const unsigned short* wptr = Wt + (size_t)(g * kHID + j0 + lr) * K;
  const int m0 = half * 32 + lr;    // row for first M tile
  const int m1 = m0 + 16;           // row for second M tile

  // ---- Phase 1: recurrent part, K = 0..1023 (32 blocks), branchless ----
  {
    const unsigned short* a0p = h_prev + (size_t)m0 * kHID;
    const unsigned short* a1p = h_prev + (size_t)m1 * kHID;
    for (int kb = 0; kb < kHID / 32; ++kb) {
      Frag b, a0, a1;
      const int kk = kb * 32 + kh * 8;
      b.q[0] = *(const u32x4*)(wptr + kk);
      b.q[1] = *(const u32x4*)(wptr + kk + 16);
      const int k0 = kb * 32 + kh * 8;
      a0.q[0] = *(const u32x4*)(a0p + k0);
      a0.q[1] = *(const u32x4*)(a0p + k0 + 16);
      a1.q[0] = *(const u32x4*)(a1p + k0);
      a1.q[1] = *(const u32x4*)(a1p + k0 + 16);
      acc0 = __builtin_amdgcn_wmma_f32_16x16x32_bf16(
          false, a0.v, false, b.v, (short)0, acc0, false, false);
      acc1 = __builtin_amdgcn_wmma_f32_16x16x32_bf16(
          false, a1.v, false, b.v, (short)0, acc1, false, false);
    }
  }

  // ---- Phase 2: input part, K = 1024..K-1, branchless ----
  {
    const unsigned short* wpx = wptr + kHID;   // K offset 1024 within the row
    const unsigned short* a0p = x + (size_t)m0 * xdim;
    const unsigned short* a1p = x + (size_t)m1 * xdim;
    const int nx = (K - kHID) >> 5;
    for (int kb = 0; kb < nx; ++kb) {
      Frag b, a0, a1;
      const int kk = kb * 32 + kh * 8;
      b.q[0] = *(const u32x4*)(wpx + kk);
      b.q[1] = *(const u32x4*)(wpx + kk + 16);
      const int k0 = kb * 32 + kh * 8;
      a0.q[0] = *(const u32x4*)(a0p + k0);
      a0.q[1] = *(const u32x4*)(a0p + k0 + 16);
      a1.q[0] = *(const u32x4*)(a1p + k0);
      a1.q[1] = *(const u32x4*)(a1p + k0 + 16);
      acc0 = __builtin_amdgcn_wmma_f32_16x16x32_bf16(
          false, a0.v, false, b.v, (short)0, acc0, false, false);
      acc1 = __builtin_amdgcn_wmma_f32_16x16x32_bf16(
          false, a1.v, false, b.v, (short)0, acc1, false, false);
    }
  }

  // Spill gate pre-activations to LDS:  C/D layout: lane lr = col, row = kh*8+v.
  {
    const int rbase = half * 32 + kh * 8;
#pragma unroll
    for (int v = 0; v < 8; ++v) {
      gbuf[g][rbase + v][lr]      = acc0[v];
      gbuf[g][rbase + 16 + v][lr] = acc1[v];
    }
  }
  __syncthreads();

  // Fused LSTM cell update for this 64x16 column slice.
  for (int idx = tid; idx < kBATCH * 16; idx += 256) {
    const int m = idx >> 4;
    const int n = idx & 15;
    const int j = j0 + n;
    const float iv = sigmoidf_(gbuf[0][m][n] + bi[j]);
    const float gv = tanhf   (gbuf[1][m][n] + bc[j]);
    const float fv = sigmoidf_(gbuf[2][m][n] + bff[j]);
    const float ov = sigmoidf_(gbuf[3][m][n] + bo[j]);
    const size_t o = (size_t)m * kHID + j;
    const float c  = fv * c_state[o] + iv * gv;
    c_state[o] = c;
    h_out[o]   = f2bf(ov * tanhf(c));
  }
}

// ---------------------------------------------------------------------------
// Final projection: logits = h1 (64x1024 bf16) @ W_out + b_out.
// grid = NCLASS/32, block = 256 (8 waves): wave = (n-subtile, m-tile).
// ---------------------------------------------------------------------------
__global__ __launch_bounds__(256) void proj_kernel(
    const unsigned short* __restrict__ WtOut,  // [NCLASS][1024] bf16
    const unsigned short* __restrict__ h,      // 64x1024 bf16
    const float* __restrict__ b_out,
    float* __restrict__ out) {
  const int tid  = threadIdx.x;
  const int wave = tid >> 5;
  const int lane = tid & 31;
  const int nt   = wave >> 2;
  const int mt   = wave & 3;
  const int lr   = lane & 15;
  const int kh   = lane >> 4;
  const int n    = blockIdx.x * 32 + nt * 16 + lr;

  v8f acc = {};
  const unsigned short* wptr = WtOut + (size_t)n * kHID;
  const unsigned short* aptr = h + (size_t)(mt * 16 + lr) * kHID;

  for (int kb = 0; kb < kHID / 32; ++kb) {
    Frag b, a;
    const int kk = kb * 32 + kh * 8;
    b.q[0] = *(const u32x4*)(wptr + kk);
    b.q[1] = *(const u32x4*)(wptr + kk + 16);
    const int k0 = kb * 32 + kh * 8;
    a.q[0] = *(const u32x4*)(aptr + k0);
    a.q[1] = *(const u32x4*)(aptr + k0 + 16);
    acc = __builtin_amdgcn_wmma_f32_16x16x32_bf16(
        false, a.v, false, b.v, (short)0, acc, false, false);
  }

  const float bias = b_out[n];
#pragma unroll
  for (int v = 0; v < 8; ++v) {
    const int m = mt * 16 + kh * 8 + v;
    out[(size_t)m * kNCLASS + n] = acc[v] + bias;
  }
}

// ---------------------------------------------------------------------------
// Host orchestration: weight/embedding prep once per call, then 256 x 2 step
// kernels, then the projection. All launches on `stream` (graph-capture safe).
// ---------------------------------------------------------------------------
extern "C" void kernel_launch(void* const* d_in, const int* in_sizes, int n_in,
                              void* d_out, int out_size, void* d_ws, size_t ws_size,
                              hipStream_t stream) {
  (void)in_sizes; (void)n_in; (void)out_size;

  const int*   X     = (const int*)d_in[0];
  const float* C_emb = (const float*)d_in[1];
  const float* W0[4] = {(const float*)d_in[2], (const float*)d_in[4],
                        (const float*)d_in[6], (const float*)d_in[8]};
  const float* B0[4] = {(const float*)d_in[3], (const float*)d_in[5],
                        (const float*)d_in[7], (const float*)d_in[9]};
  const float* W1[4] = {(const float*)d_in[10], (const float*)d_in[12],
                        (const float*)d_in[14], (const float*)d_in[16]};
  const float* B1[4] = {(const float*)d_in[11], (const float*)d_in[13],
                        (const float*)d_in[15], (const float*)d_in[17]};
  const float* W_out = (const float*)d_in[18];
  const float* b_out = (const float*)d_in[19];
  float*       out   = (float*)d_out;

  const int K0 = kHID + kEMB;   // 1536
  const int K1 = kHID + kHID;   // 2048
  const int HB = kBATCH * kHID; // elements per h/c buffer

  uint8_t* p = (uint8_t*)d_ws;
  auto take = [&](size_t bytes) -> uint8_t* {
    uint8_t* r = p;
    p += (bytes + 255) & ~(size_t)255;
    return r;
  };
  unsigned short* Wt0   = (unsigned short*)take(4ull * kHID * K0 * 2);
  unsigned short* Wt1   = (unsigned short*)take(4ull * kHID * K1 * 2);
  unsigned short* WtOut = (unsigned short*)take((size_t)kNCLASS * kHID * 2);
  unsigned short* xs    = (unsigned short*)take((size_t)kSEQ * kBATCH * kEMB * 2);
  unsigned short* h0    = (unsigned short*)take(2ull * HB * 2);
  unsigned short* h1    = (unsigned short*)take(2ull * HB * 2);
  float*          c0    = (float*)take((size_t)HB * 4);
  float*          c1    = (float*)take((size_t)HB * 4);
  if ((size_t)(p - (uint8_t*)d_ws) > ws_size) return;  // insufficient scratch

  init_state<<<dim3((HB + 255) / 256), dim3(256), 0, stream>>>(
      h0 + HB, h1 + HB, c0, c1);

  embed_gather_bf16<<<dim3(kSEQ * kBATCH), dim3(128), 0, stream>>>(
      C_emb, X, xs);

  const dim3 tb(32, 8);
  for (int g = 0; g < 4; ++g) {
    transpose_f32_to_bf16<<<dim3(kHID / 32, K0 / 32), tb, 0, stream>>>(
        W0[g], Wt0 + (size_t)g * kHID * K0, K0, kHID);
    transpose_f32_to_bf16<<<dim3(kHID / 32, K1 / 32), tb, 0, stream>>>(
        W1[g], Wt1 + (size_t)g * kHID * K1, K1, kHID);
  }
  transpose_f32_to_bf16<<<dim3(kNCLASS / 32, kHID / 32), tb, 0, stream>>>(
      W_out, WtOut, kHID, kNCLASS);

  for (int t = 0; t < kSEQ; ++t) {
    const int par = t & 1;
    // layer 0: A = cat(h0_prev, xs[t])
    lstm_step_kernel<<<kHID / 16, 256, 0, stream>>>(
        Wt0, K0, h0 + (par ^ 1) * HB,
        xs + (size_t)t * kBATCH * kEMB, kEMB,
        B0[0], B0[1], B0[2], B0[3], c0, h0 + par * HB);
    // layer 1: A = cat(h1_prev, h0_cur)
    lstm_step_kernel<<<kHID / 16, 256, 0, stream>>>(
        Wt1, K1, h1 + (par ^ 1) * HB,
        h0 + par * HB, kHID,
        B1[0], B1[1], B1[2], B1[3], c1, h1 + par * HB);
  }

  // t = 255 has parity 1 -> final h1 lives in buffer 1.
  proj_kernel<<<kNCLASS / 32, 256, 0, stream>>>(WtOut, h1 + HB, b_out, out);
}
